// NeuralODEFunc_953482740302
// MI455X (gfx1250) — matrix operand order repaired and checked by reading, not compiled, not run on hardware
//
#include <hip/hip_runtime.h>
#include <stdint.h>

typedef __bf16 bf16;
typedef __attribute__((ext_vector_type(16))) __bf16 v16bf;
typedef __attribute__((ext_vector_type(8)))  float  v8f;

// ---------------------------------------------------------------------------
// WMMA helpers (bf16 A/B, f32 accumulate), CDNA5 gfx1250 layouts (ISA 7.12.2)
// ---------------------------------------------------------------------------
__device__ __forceinline__ v8f wmma_bf16(v16bf a, v16bf b, v8f c) {
  return __builtin_amdgcn_wmma_f32_16x16x32_bf16(false, a, false, b, (short)0, c,
                                                 false, false);
}

// A fragment (16x32 bf16) from an LDS row (row-major panel, 4B-aligned pitch).
// lane<16: row=lane,    K = {0..7, 16..23}
// lane>=16: row=lane-16, K = {8..15, 24..31}
__device__ __forceinline__ v16bf a_frag(const bf16* row, int kbase, int lhi) {
  const uint32_t* q = (const uint32_t*)(row + kbase + (lhi ? 8 : 0));
  union { uint32_t u[8]; v16bf v; } r;
  r.u[0] = q[0];  r.u[1] = q[1];  r.u[2] = q[2];  r.u[3] = q[3];
  r.u[4] = q[8];  r.u[5] = q[9];  r.u[6] = q[10]; r.u[7] = q[11];
  return r.v;
}

// B fragment: pre-swizzled storage, 16 contiguous bf16 per lane per tile.
__device__ __forceinline__ v16bf b_frag(const bf16* frags, int fragIdx, int lane) {
  return *(const v16bf*)(frags + (((size_t)fragIdx * 32 + lane) << 4));
}

// Order-preserving float <-> uint encoding for atomicMax-based segment max.
__device__ __forceinline__ unsigned fenc(float f) {
  unsigned u = __float_as_uint(f);
  return (u & 0x80000000u) ? ~u : (u | 0x80000000u);
}
__device__ __forceinline__ float fdec(unsigned u) {
  return (u & 0x80000000u) ? __uint_as_float(u & 0x7fffffffu)
                           : __uint_as_float(~u);
}

__device__ __forceinline__ float silu(float v) { return v / (1.f + __expf(-v)); }

// ---------------------------------------------------------------------------
// Kernel: repack fp32 weight [K,N] into bf16 B-fragment order.
// Fragment (kt,nt): lane l<16 -> col nt*16+l, K = kt*32 + {0..7, 16..23}
//                   lane l>=16 -> col nt*16+(l-16), K = kt*32 + {8..15, 24..31}
// ---------------------------------------------------------------------------
__global__ __launch_bounds__(256) void wconv_kernel(const float* __restrict__ w,
                                                    bf16* __restrict__ out,
                                                    int K, int Ncols, int total) {
  int i = blockIdx.x * 256 + threadIdx.x;
  if (i >= total) return;
  int k = i / Ncols, n = i - k * Ncols;
  int kt = k >> 5, kin = k & 31, nt = n >> 4, nin = n & 15;
  int lane = nin + ((kin & 8) ? 16 : 0);
  int elem = (kin & 7) | ((kin >> 1) & 8);
  int ntiles = Ncols >> 4;
  out[(((size_t)(kt * ntiles + nt)) * 32 + lane) * 16 + elem] = (bf16)w[i];
}

// ---------------------------------------------------------------------------
// Kernel: per-node projections. hb = bf16(h); qh = h@wq; kh = h@wk.
// 4 waves/block, each wave computes a 16-node x 128-col tile with WMMA.
// ---------------------------------------------------------------------------
__global__ __launch_bounds__(128) void node_proj_kernel(
    const float* __restrict__ h, const bf16* __restrict__ wqf,
    const bf16* __restrict__ wkf, bf16* __restrict__ hb,
    float* __restrict__ qh, float* __restrict__ kh, int N) {
  __shared__ bf16 sA[4][16 * 132];
  int wv = threadIdx.x >> 5, lane = threadIdx.x & 31;
  int base = blockIdx.x * 64 + wv * 16;
  int r = lane & 15, lhi = lane >> 4;
  int node = base + r;
  bool ok = node < N;
  {
    const float* hrow = h + (size_t)node * 128;
    bf16* arow = &sA[wv][r * 132];
    bf16* grow = hb + (size_t)node * 128;
#pragma unroll
    for (int c4 = 0; c4 < 16; c4++) {
      int c = lhi * 64 + c4 * 4;
      float4 v = ok ? *(const float4*)(hrow + c) : make_float4(0.f, 0.f, 0.f, 0.f);
      bf16 x0 = (bf16)v.x, x1 = (bf16)v.y, x2 = (bf16)v.z, x3 = (bf16)v.w;
      arow[c] = x0; arow[c + 1] = x1; arow[c + 2] = x2; arow[c + 3] = x3;
      if (ok) { grow[c] = x0; grow[c + 1] = x1; grow[c + 2] = x2; grow[c + 3] = x3; }
    }
  }
  __syncthreads();
  v8f aq[8] = {}, ak[8] = {};
  const bf16* Arow = &sA[wv][r * 132];
#pragma unroll
  for (int kt = 0; kt < 4; kt++) {
    v16bf a = a_frag(Arow, kt * 32, lhi);
#pragma unroll
    for (int nt = 0; nt < 8; nt++) {
      aq[nt] = wmma_bf16(a, b_frag(wqf, kt * 8 + nt, lane), aq[nt]);
      ak[nt] = wmma_bf16(a, b_frag(wkf, kt * 8 + nt, lane), ak[nt]);
    }
  }
#pragma unroll
  for (int nt = 0; nt < 8; nt++) {
    int col = nt * 16 + r;
#pragma unroll
    for (int rr = 0; rr < 8; rr++) {
      int nd = base + rr + lhi * 8;
      if (nd < N) {
        qh[(size_t)nd * 128 + col] = aq[nt][rr];
        kh[(size_t)nd * 128 + col] = ak[nt][rr];
      }
    }
  }
}

// ---------------------------------------------------------------------------
// Kernel: attention scores per (edge, head) + segment max via atomicMax.
// score = q[tgt] . (kh[src] + dt_feat@wt) / sqrt(DH)
// ---------------------------------------------------------------------------
__global__ __launch_bounds__(256) void edge_scores_kernel(
    const int* __restrict__ ei, const float* __restrict__ dt,
    const float* __restrict__ freq, const float* __restrict__ phase,
    const float* __restrict__ wt, const float* __restrict__ qh,
    const float* __restrict__ kh, float* __restrict__ scores,
    unsigned* __restrict__ smax, int E) {
  __shared__ float s_wt[32 * 128];
  __shared__ float s_fr[32], s_ph[32];
  for (int i = threadIdx.x; i < 32 * 128; i += 256) s_wt[i] = wt[i];
  if (threadIdx.x < 32) {
    s_fr[threadIdx.x] = freq[threadIdx.x];
    s_ph[threadIdx.x] = phase[threadIdx.x];
  }
  __syncthreads();
  int tId = blockIdx.x * 256 + threadIdx.x;
  int e = tId >> 2, hd = tId & 3;
  if (e >= E) return;
  float d0 = dt[e];
  float tf[32];
#pragma unroll
  for (int j = 0; j < 32; j++) tf[j] = __cosf(d0 * s_fr[j] + s_ph[j]);
  int tg = ei[e], sr = ei[E + e];
  const float* qr = qh + (size_t)tg * 128 + hd * 32;
  const float* kr = kh + (size_t)sr * 128 + hd * 32;
  const float* wcol = &s_wt[hd * 32];
  float sc = 0.f;
  for (int d = 0; d < 32; d++) {
    float acc = 0.f;
#pragma unroll
    for (int j = 0; j < 32; j++) acc = fmaf(tf[j], wcol[j * 128 + d], acc);
    sc = fmaf(qr[d], kr[d] + acc, sc);
  }
  sc *= 0.17677669529663688f;  // 1/sqrt(32)
  scores[(size_t)e * 4 + hd] = sc;
  atomicMax(&smax[(size_t)tg * 4 + hd], fenc(sc));
}

// ---------------------------------------------------------------------------
// Kernel: ex = exp(score - segmax); den += ex (segment sum).
// ---------------------------------------------------------------------------
__global__ __launch_bounds__(256) void edge_softmax_kernel(
    const int* __restrict__ ei, float* __restrict__ scores,
    const unsigned* __restrict__ smax, float* __restrict__ den, int E) {
  int tId = blockIdx.x * 256 + threadIdx.x;
  if (tId >= E * 4) return;
  int e = tId >> 2, hd = tId & 3;
  int tg = ei[e];
  float m = fdec(smax[(size_t)tg * 4 + hd]);
  float ex = __expf(scores[tId] - m);
  scores[tId] = ex;
  atomicAdd(&den[(size_t)tg * 4 + hd], ex);
}

// ---------------------------------------------------------------------------
// Kernel: message MLP (WMMA) + attention-weighted scatter into agg.
// 4 waves/block, each wave handles 16 edges (full 128-wide output).
// ---------------------------------------------------------------------------
__global__ __launch_bounds__(128) void edge_msg_kernel(
    const int* __restrict__ ei, const bf16* __restrict__ hb,
    const float* __restrict__ efeat, const bf16* __restrict__ w1f,
    const bf16* __restrict__ w2f, const float* __restrict__ b1v,
    const float* __restrict__ b2v, const float* __restrict__ exs,
    const float* __restrict__ den, float* __restrict__ agg, int E) {
  __shared__ bf16 sA[4][16 * 168];   // [16 edges][160 K + pad]
  __shared__ bf16 sH[4][16 * 136];   // hidden [16][128 + pad]
  __shared__ float sAttn[4][64];     // [16 edges][4 heads]
  int wv = threadIdx.x >> 5, lane = threadIdx.x & 31;
  int ebase = blockIdx.x * 64 + wv * 16;
  int r = lane & 15, lhi = lane >> 4;
  {
    int e = ebase + r;
    bool ok = e < E;
    int sr = ok ? ei[E + e] : 0;
    const uint32_t* hrow = (const uint32_t*)(hb + (size_t)sr * 128);
    uint32_t* arow = (uint32_t*)&sA[wv][r * 168];
#pragma unroll
    for (int c = 0; c < 32; c++) arow[lhi * 32 + c] = ok ? hrow[lhi * 32 + c] : 0u;
    const float* fe = efeat + (size_t)e * 32;
#pragma unroll
    for (int c = 0; c < 16; c++) {
      int fc = lhi * 16 + c;
      sA[wv][r * 168 + 128 + fc] = (bf16)(ok ? fe[fc] : 0.f);
    }
  }
  for (int idx = lane; idx < 64; idx += 32) {
    int rr = idx >> 2, hd = idx & 3;
    int e = ebase + rr;
    float a = 0.f;
    if (e < E) {
      int tg = ei[e];
      float dn = den[(size_t)tg * 4 + hd];
      a = exs[(size_t)e * 4 + hd] / fmaxf(dn, 1e-30f);
    }
    sAttn[wv][idx] = a;
  }
  __syncthreads();
  // GEMM1: [16x160] @ [160x128]
  v8f acc[8] = {};
  const bf16* Arow = &sA[wv][r * 168];
#pragma unroll
  for (int kt = 0; kt < 5; kt++) {
    v16bf a = a_frag(Arow, kt * 32, lhi);
#pragma unroll
    for (int nt = 0; nt < 8; nt++)
      acc[nt] = wmma_bf16(a, b_frag(w1f, kt * 8 + nt, lane), acc[nt]);
  }
#pragma unroll
  for (int nt = 0; nt < 8; nt++) {
    int col = nt * 16 + r;
    float bias = b1v[col];
#pragma unroll
    for (int rr = 0; rr < 8; rr++)
      sH[wv][(rr + lhi * 8) * 136 + col] = (bf16)silu(acc[nt][rr] + bias);
  }
  __syncthreads();
  // GEMM2: [16x128] @ [128x128]
  v8f acc2[8] = {};
  const bf16* Hrow = &sH[wv][r * 136];
#pragma unroll
  for (int kt = 0; kt < 4; kt++) {
    v16bf a = a_frag(Hrow, kt * 32, lhi);
#pragma unroll
    for (int nt = 0; nt < 8; nt++)
      acc2[nt] = wmma_bf16(a, b_frag(w2f, kt * 8 + nt, lane), acc2[nt]);
  }
  int tgs[8]; bool oks[8];
#pragma unroll
  for (int rr = 0; rr < 8; rr++) {
    int e = ebase + rr + lhi * 8;
    oks[rr] = e < E;
    tgs[rr] = oks[rr] ? ei[e] : 0;
  }
#pragma unroll
  for (int nt = 0; nt < 8; nt++) {
    int col = nt * 16 + r;
    int hd = col >> 5;
    float bias = b2v[col];
#pragma unroll
    for (int rr = 0; rr < 8; rr++) {
      if (oks[rr]) {
        float v = (acc2[nt][rr] + bias) * sAttn[wv][(rr + lhi * 8) * 4 + hd];
        atomicAdd(&agg[(size_t)tgs[rr] * 128 + col], v);
      }
    }
  }
}

// ---------------------------------------------------------------------------
// Kernel: aggregated = agg@wo; x=[h||aggregated||t_feat]; 3-layer MLP -> dh.
// 2 waves/block (keeps static LDS < 64KB), 16 nodes per wave, all WMMA.
// ---------------------------------------------------------------------------
__global__ __launch_bounds__(64) void node_out_kernel(
    const float* __restrict__ h, const float* __restrict__ agg,
    const float* __restrict__ tptr, const float* __restrict__ freq,
    const float* __restrict__ phase, const bf16* __restrict__ wof,
    const bf16* __restrict__ w1f, const bf16* __restrict__ w2f,
    const bf16* __restrict__ w3f, const float* __restrict__ nb1,
    const float* __restrict__ nb2, const float* __restrict__ nb3,
    float* __restrict__ out, int N) {
  __shared__ bf16 sX[2][16 * 296];  // x panel [16][288+pad]; reused as H2 [16][264]
  __shared__ bf16 sH[2][16 * 264];  // tmpA [16][132] first, then hidden1 [16][264]
  int wv = threadIdx.x >> 5, lane = threadIdx.x & 31;
  int base = blockIdx.x * 32 + wv * 16;
  int r = lane & 15, lhi = lane >> 4;
  int node = base + r;
  bool ok = node < N;
  {
    const float* arow = agg + (size_t)node * 128;
    const float* hrow = h + (size_t)node * 128;
    bf16* trow = &sH[wv][r * 132];
    bf16* xrow = &sX[wv][r * 296];
#pragma unroll
    for (int c = 0; c < 64; c++) {
      int cc = lhi * 64 + c;
      trow[cc] = (bf16)(ok ? arow[cc] : 0.f);
      xrow[cc] = (bf16)(ok ? hrow[cc] : 0.f);
    }
    float t0 = tptr[0];
    float tv = __cosf(t0 * freq[lane] + phase[lane]);  // node-independent
#pragma unroll
    for (int rr = 0; rr < 16; rr++) sX[wv][rr * 296 + 256 + lane] = (bf16)tv;
  }
  __syncthreads();
  // aggregated = agg @ wo
  v8f accO[8] = {};
  {
    const bf16* Arow = &sH[wv][r * 132];
#pragma unroll
    for (int kt = 0; kt < 4; kt++) {
      v16bf a = a_frag(Arow, kt * 32, lhi);
#pragma unroll
      for (int nt = 0; nt < 8; nt++)
        accO[nt] = wmma_bf16(a, b_frag(wof, kt * 8 + nt, lane), accO[nt]);
    }
  }
  __syncthreads();
#pragma unroll
  for (int nt = 0; nt < 8; nt++) {
    int col = 128 + nt * 16 + r;
#pragma unroll
    for (int rr = 0; rr < 8; rr++)
      sX[wv][(rr + lhi * 8) * 296 + col] = (bf16)accO[nt][rr];
  }
  __syncthreads();
  // layer 1: [16x288] @ [288x256]
  v8f acc1[16] = {};
  {
    const bf16* Xrow = &sX[wv][r * 296];
    for (int kt = 0; kt < 9; kt++) {
      v16bf a = a_frag(Xrow, kt * 32, lhi);
#pragma unroll
      for (int nt = 0; nt < 16; nt++)
        acc1[nt] = wmma_bf16(a, b_frag(w1f, kt * 16 + nt, lane), acc1[nt]);
    }
  }
  __syncthreads();
#pragma unroll
  for (int nt = 0; nt < 16; nt++) {
    int col = nt * 16 + r;
    float bias = nb1[col];
#pragma unroll
    for (int rr = 0; rr < 8; rr++)
      sH[wv][(rr + lhi * 8) * 264 + col] = (bf16)silu(acc1[nt][rr] + bias);
  }
  __syncthreads();
  // layer 2: [16x256] @ [256x256]
  v8f acc2[16] = {};
  {
    const bf16* Hrow = &sH[wv][r * 264];
    for (int kt = 0; kt < 8; kt++) {
      v16bf a = a_frag(Hrow, kt * 32, lhi);
#pragma unroll
      for (int nt = 0; nt < 16; nt++)
        acc2[nt] = wmma_bf16(a, b_frag(w2f, kt * 16 + nt, lane), acc2[nt]);
    }
  }
  __syncthreads();
#pragma unroll
  for (int nt = 0; nt < 16; nt++) {
    int col = nt * 16 + r;
    float bias = nb2[col];
#pragma unroll
    for (int rr = 0; rr < 8; rr++)
      sX[wv][(rr + lhi * 8) * 264 + col] = (bf16)silu(acc2[nt][rr] + bias);
  }
  __syncthreads();
  // layer 3: [16x256] @ [256x128]
  v8f acc3[8] = {};
  {
    const bf16* H2row = &sX[wv][r * 264];
#pragma unroll
    for (int kt = 0; kt < 8; kt++) {
      v16bf a = a_frag(H2row, kt * 32, lhi);
#pragma unroll
      for (int nt = 0; nt < 8; nt++)
        acc3[nt] = wmma_bf16(a, b_frag(w3f, kt * 8 + nt, lane), acc3[nt]);
    }
  }
#pragma unroll
  for (int nt = 0; nt < 8; nt++) {
    int col = nt * 16 + r;
    float bias = nb3[col];
#pragma unroll
    for (int rr = 0; rr < 8; rr++) {
      int nd = base + rr + lhi * 8;
      if (nd < N) out[(size_t)nd * 128 + col] = acc3[nt][rr] + bias;
    }
  }
}

// ---------------------------------------------------------------------------
// Host launcher
// ---------------------------------------------------------------------------
extern "C" void kernel_launch(void* const* d_in, const int* in_sizes, int n_in,
                              void* d_out, int out_size, void* d_ws, size_t ws_size,
                              hipStream_t stream) {
  const float* t     = (const float*)d_in[0];
  const float* h     = (const float*)d_in[1];
  const int*   ei    = (const int*)d_in[2];
  const float* efeat = (const float*)d_in[3];
  const float* dt    = (const float*)d_in[4];
  const float* mw1   = (const float*)d_in[6];
  const float* mb1   = (const float*)d_in[7];
  const float* mw2   = (const float*)d_in[8];
  const float* mb2   = (const float*)d_in[9];
  const float* wq    = (const float*)d_in[10];
  const float* wk    = (const float*)d_in[11];
  const float* wt    = (const float*)d_in[12];
  const float* wo    = (const float*)d_in[13];
  const float* tfr   = (const float*)d_in[14];
  const float* tph   = (const float*)d_in[15];
  const float* nw1   = (const float*)d_in[16];
  const float* nb1   = (const float*)d_in[17];
  const float* nw2   = (const float*)d_in[18];
  const float* nb2   = (const float*)d_in[19];
  const float* nw3   = (const float*)d_in[20];
  const float* nb3   = (const float*)d_in[21];
  const int N = in_sizes[1] / 128;
  const int E = in_sizes[4];
  (void)n_in; (void)out_size; (void)ws_size;

  char* ws = (char*)d_ws;
  size_t off = 0;
  auto take = [&](size_t bytes) {
    char* p = ws + off;
    off = (off + bytes + 255) & ~(size_t)255;
    return p;
  };
  unsigned* smax = (unsigned*)take((size_t)N * 4 * sizeof(unsigned));
  float* den     = (float*)take((size_t)N * 4 * sizeof(float));
  float* agg     = (float*)take((size_t)N * 128 * sizeof(float));
  size_t zeroBytes = off;  // smax | den | agg contiguous from ws start
  bf16* hb       = (bf16*)take((size_t)N * 128 * sizeof(bf16));
  float* qh      = (float*)take((size_t)N * 128 * sizeof(float));
  float* kh      = (float*)take((size_t)N * 128 * sizeof(float));
  float* scores  = (float*)take((size_t)E * 4 * sizeof(float));
  bf16* wqf  = (bf16*)take((size_t)128 * 128 * 2);
  bf16* wkf  = (bf16*)take((size_t)128 * 128 * 2);
  bf16* wof  = (bf16*)take((size_t)128 * 128 * 2);
  bf16* mw1f = (bf16*)take((size_t)160 * 128 * 2);
  bf16* mw2f = (bf16*)take((size_t)128 * 128 * 2);
  bf16* n1f  = (bf16*)take((size_t)288 * 256 * 2);
  bf16* n2f  = (bf16*)take((size_t)256 * 256 * 2);
  bf16* n3f  = (bf16*)take((size_t)256 * 128 * 2);

  hipMemsetAsync(d_ws, 0, zeroBytes, stream);

  auto conv = [&](const float* w, bf16* f, int K, int Ncols) {
    int tot = K * Ncols;
    wconv_kernel<<<dim3((tot + 255) / 256), dim3(256), 0, stream>>>(w, f, K, Ncols, tot);
  };
  conv(wq, wqf, 128, 128);
  conv(wk, wkf, 128, 128);
  conv(wo, wof, 128, 128);
  conv(mw1, mw1f, 160, 128);
  conv(mw2, mw2f, 128, 128);
  conv(nw1, n1f, 288, 256);
  conv(nw2, n2f, 256, 256);
  conv(nw3, n3f, 256, 128);

  node_proj_kernel<<<dim3((N + 63) / 64), dim3(128), 0, stream>>>(h, wqf, wkf, hb,
                                                                  qh, kh, N);
  int eth = E * 4;
  edge_scores_kernel<<<dim3((eth + 255) / 256), dim3(256), 0, stream>>>(
      ei, dt, tfr, tph, wt, qh, kh, scores, smax, E);
  edge_softmax_kernel<<<dim3((eth + 255) / 256), dim3(256), 0, stream>>>(
      ei, scores, smax, den, E);
  edge_msg_kernel<<<dim3((E + 63) / 64), dim3(128), 0, stream>>>(
      ei, hb, efeat, mw1f, mw2f, mb1, mb2, scores, den, agg, E);
  node_out_kernel<<<dim3((N + 31) / 32), dim3(64), 0, stream>>>(
      h, agg, t, tfr, tph, wof, n1f, n2f, n3f, nb1, nb2, nb3, (float*)d_out, N);
}
